// PGCTDecoder_61134564491393
// MI455X (gfx1250) — compile-verified, matrix operand order checked by reading, not run on hardware
//
#include <hip/hip_runtime.h>

// ---------------------------------------------------------------------------
// MI455X (gfx1250, wave32) pointer-generator decoder.
// GEMMs: v_wmma_f32_16x16x32_f16, 128x128x32 block tiles, 8 waves/block,
// each wave 2x4 accumulators (8 WMMA per K-step, A/B fragment reuse).
// enc_feat GEMM stages tiles through the Tensor Data Mover (tensor_load_to_lds
// + s_wait_tensorcnt) with TDM pad fields reproducing the LDS row pitch.
// ---------------------------------------------------------------------------

typedef _Float16 f16;
typedef __attribute__((ext_vector_type(16))) _Float16 v16h;
typedef __attribute__((ext_vector_type(8)))  _Float16 v8h;
typedef __attribute__((ext_vector_type(8)))  float    v8f;
typedef __attribute__((ext_vector_type(4)))  unsigned int u32x4;
typedef __attribute__((ext_vector_type(4)))  int      i32x4;
typedef __attribute__((ext_vector_type(8)))  int      i32x8;

#define D_MODEL 512
#define N_HEAD  8
#define D_HEAD  64
#define B_SZ    16
#define T_FULL  100
#define TI      99
#define S_LEN   400
#define V_SZ    32000
#define M_ROWS  (B_SZ * TI)   // 1584

#if __has_builtin(__builtin_amdgcn_tensor_load_to_lds) && \
    __has_builtin(__builtin_amdgcn_s_wait_tensorcnt)
#define HAVE_TDM 1
#else
#define HAVE_TDM 0
#endif

// ---------------------------------------------------------------------------
// Elementwise helpers
// ---------------------------------------------------------------------------
__global__ void k_convert(const float* __restrict__ in, f16* __restrict__ out, long n) {
  long i = (long)blockIdx.x * blockDim.x + threadIdx.x;
  long stride = (long)gridDim.x * blockDim.x;
  for (; i < n; i += stride) out[i] = (f16)in[i];
}

__global__ void k_zero(float* __restrict__ p, long n) {
  long i = (long)blockIdx.x * blockDim.x + threadIdx.x;
  long stride = (long)gridDim.x * blockDim.x;
  for (; i < n; i += stride) p[i] = 0.f;
}

// ---------------------------------------------------------------------------
// Fragment gather from LDS per CDNA5 ISA 7.12.2 (16-bit A 16x32 layout):
// lanes 0-15 -> K-base 0, lanes 16-31 -> K-base 8; halves j -> K pairs.
// ---------------------------------------------------------------------------
#define LOAD_FRAG(dst, ARR, r, kb)                                   \
  do {                                                               \
    for (int j = 0; j < 4; ++j) {                                    \
      dst[2*j]     = ARR[r][(kb) + 2*j];                             \
      dst[2*j+1]   = ARR[r][(kb) + 2*j + 1];                         \
      dst[8+2*j]   = ARR[r][16 + (kb) + 2*j];                        \
      dst[8+2*j+1] = ARR[r][16 + (kb) + 2*j + 1];                    \
    }                                                                \
  } while (0)

// ---------------------------------------------------------------------------
// WMMA GEMM: C[M,N] = A[M,K](f16) @ W[N,K]^T(f16) (+bias[n]) (+res) (relu)
// 256 threads = 8 wave32 waves; block tile 128x128, K-step 32.
// Wave w: M-offset (w&3)*32 (2 subtiles), N-offset (w>>2)*64 (4 subtiles).
// ---------------------------------------------------------------------------
__global__ __launch_bounds__(256) void k_gemm(
    const f16* __restrict__ A, const f16* __restrict__ W,
    const float* __restrict__ bias, const float* __restrict__ res,
    float* __restrict__ Cf, f16* __restrict__ Ch,
    int M, int N, int K, int relu)
{
  __shared__ f16 As[128][40];   // 32 halves + 8 pad -> conflict-free
  __shared__ f16 Bs[128][40];

  const int tid  = threadIdx.x;
  const int n0   = blockIdx.x * 128;
  const int m0   = blockIdx.y * 128;
  const int wave = tid >> 5;
  const int lane = tid & 31;
  const int wm   = (wave & 3) * 32;
  const int wn   = (wave >> 2) * 64;

  const v8f vzero = {0.f,0.f,0.f,0.f,0.f,0.f,0.f,0.f};
  v8f acc[2][4];
  #pragma unroll
  for (int i = 0; i < 2; ++i)
    #pragma unroll
    for (int j = 0; j < 4; ++j) acc[i][j] = vzero;

  const int srow = tid >> 1;           // staging row 0..127
  const int scol = (tid & 1) * 16;     // staging col 0/16 (two v8h chunks)

  for (int k0 = 0; k0 < K; k0 += 32) {
    __syncthreads();
    {
      const long grow = m0 + srow;
      v8h a0 = {0,0,0,0,0,0,0,0}, a1 = {0,0,0,0,0,0,0,0};
      if (grow < M) {
        const f16* ap = A + grow * (long)K + k0 + scol;
        a0 = *(const v8h*)ap;
        a1 = *(const v8h*)(ap + 8);
      }
      #pragma unroll
      for (int i = 0; i < 8; ++i) { As[srow][scol+i] = a0[i]; As[srow][scol+8+i] = a1[i]; }

      const long wrow = n0 + srow;
      v8h b0 = {0,0,0,0,0,0,0,0}, b1 = {0,0,0,0,0,0,0,0};
      if (wrow < N) {
        const f16* wp = W + wrow * (long)K + k0 + scol;
        b0 = *(const v8h*)wp;
        b1 = *(const v8h*)(wp + 8);
      }
      #pragma unroll
      for (int i = 0; i < 8; ++i) { Bs[srow][scol+i] = b0[i]; Bs[srow][scol+8+i] = b1[i]; }
    }
    if (k0 + 32 < K) {   // next K tile -> global_prefetch_b8
      __builtin_prefetch(A + (long)(m0 + srow) * K + k0 + 32 + scol, 0, 0);
      __builtin_prefetch(W + (long)(n0 + srow) * K + k0 + 32 + scol, 0, 0);
    }
    __syncthreads();

    const int fr = lane & 15;
    const int kb = (lane >> 4) * 8;
    v16h af0, af1, bf0, bf1, bf2, bf3;
    LOAD_FRAG(af0, As, wm + fr,        kb);
    LOAD_FRAG(af1, As, wm + 16 + fr,   kb);
    LOAD_FRAG(bf0, Bs, wn + fr,        kb);
    LOAD_FRAG(bf1, Bs, wn + 16 + fr,   kb);
    LOAD_FRAG(bf2, Bs, wn + 32 + fr,   kb);
    LOAD_FRAG(bf3, Bs, wn + 48 + fr,   kb);

    acc[0][0] = __builtin_amdgcn_wmma_f32_16x16x32_f16(false, af0, false, bf0, (short)0, acc[0][0], false, false);
    acc[0][1] = __builtin_amdgcn_wmma_f32_16x16x32_f16(false, af0, false, bf1, (short)0, acc[0][1], false, false);
    acc[0][2] = __builtin_amdgcn_wmma_f32_16x16x32_f16(false, af0, false, bf2, (short)0, acc[0][2], false, false);
    acc[0][3] = __builtin_amdgcn_wmma_f32_16x16x32_f16(false, af0, false, bf3, (short)0, acc[0][3], false, false);
    acc[1][0] = __builtin_amdgcn_wmma_f32_16x16x32_f16(false, af1, false, bf0, (short)0, acc[1][0], false, false);
    acc[1][1] = __builtin_amdgcn_wmma_f32_16x16x32_f16(false, af1, false, bf1, (short)0, acc[1][1], false, false);
    acc[1][2] = __builtin_amdgcn_wmma_f32_16x16x32_f16(false, af1, false, bf2, (short)0, acc[1][2], false, false);
    acc[1][3] = __builtin_amdgcn_wmma_f32_16x16x32_f16(false, af1, false, bf3, (short)0, acc[1][3], false, false);
  }

  // Epilogue. C/D: VGPR e -> M = e (lanes 0-15) / 8+e (lanes 16-31), N = lane&15
  const int nf = lane & 15;
  const int mh = (lane >> 4) << 3;
  #pragma unroll
  for (int i = 0; i < 2; ++i) {
    #pragma unroll
    for (int j = 0; j < 4; ++j) {
      const int n = n0 + wn + j * 16 + nf;
      const int rowb = m0 + wm + i * 16 + mh;
      #pragma unroll
      for (int e = 0; e < 8; ++e) {
        const int row = rowb + e;
        if (row < M) {
          float v = acc[i][j][e];
          if (bias) v += bias[n];
          if (res)  v += res[(long)row * N + n];
          if (relu) v = fmaxf(v, 0.f);
          if (Cf) Cf[(long)row * N + n] = v;
          if (Ch) Ch[(long)row * N + n] = (f16)v;
        }
      }
    }
  }
}

// ---------------------------------------------------------------------------
// TDM descriptor build + issue: 2D tile (tile_k x tile_rows) of f16,
// row pitch in LDS padded by 4 DWORDs every 16 DWORDs (matches [128][40]).
// ---------------------------------------------------------------------------
#if HAVE_TDM
__device__ __forceinline__ void tdm_load_2d(
    unsigned lds_off, const f16* gbase, long row0, long k0,
    long rows_total, long K, int tile_rows, int tile_k)
{
  const unsigned long long ga = (unsigned long long)(gbase + row0 * K + k0);
  u32x4 g0;
  g0[0] = 1u;                                        // count=1 (valid, user)
  g0[1] = lds_off;                                   // lds_addr
  g0[2] = (unsigned)(ga & 0xffffffffu);              // global_addr[31:0]
  g0[3] = (unsigned)((ga >> 32) & 0x1ffffffu)        // global_addr[56:32]
        | (2u << 30);                                // type=2 ("image")
  const unsigned td0 = (unsigned)(K - k0);           // remaining K (OOB guard)
  long rr = rows_total - row0; if (rr < 0) rr = 0;
  const unsigned td1 = (unsigned)rr;                 // remaining rows
  const unsigned long long st = (unsigned long long)K;
  i32x8 g1;
  g1[0] = (int)((1u << 16)                           // data_size = 2 bytes
              | (1u << 20)                           // pad_enable
              | (3u << 22)                           // pad_interval = 16 dwords
              | (3u << 25));                         // pad_amount   = 4 dwords
  g1[1] = (int)((td0 & 0xffffu) << 16);              // tensor_dim0[15:0]
  g1[2] = (int)((td0 >> 16) | ((td1 & 0xffffu) << 16));
  g1[3] = (int)((td1 >> 16) | ((unsigned)tile_k << 16));   // tile_dim0
  g1[4] = (int)(unsigned)tile_rows;                  // tile_dim1 (tile_dim2=0)
  g1[5] = (int)(st & 0xffffffffu);                   // tensor_dim0_stride lo
  g1[6] = (int)((st >> 32) & 0xffffu);               // stride hi (dim1 stride=0)
  g1[7] = 0;
  const i32x4 z4 = {0, 0, 0, 0};
#if defined(__clang_major__) && (__clang_major__ >= 23)
  const i32x8 z8 = {0, 0, 0, 0, 0, 0, 0, 0};
  __builtin_amdgcn_tensor_load_to_lds(g0, g1, z4, z4, z8, 0);
#else
  __builtin_amdgcn_tensor_load_to_lds(g0, g1, z4, z4, 0);
#endif
}
#endif

// ---------------------------------------------------------------------------
// Same GEMM, tiles staged by the Tensor Data Mover (wave 0 issues 2 D#s,
// waits on TENSORcnt, workgroup barrier releases consumers).
// ---------------------------------------------------------------------------
__global__ __launch_bounds__(256) void k_gemm_tdm(
    const f16* __restrict__ A, const f16* __restrict__ W,
    const float* __restrict__ bias, const float* __restrict__ res,
    float* __restrict__ Cf, f16* __restrict__ Ch,
    int M, int N, int K, int relu)
{
  __shared__ f16 As[128][40];
  __shared__ f16 Bs[128][40];

  const int tid  = threadIdx.x;
  const int n0   = blockIdx.x * 128;
  const int m0   = blockIdx.y * 128;
  const int wave = tid >> 5;
  const int lane = tid & 31;
  const int wm   = (wave & 3) * 32;
  const int wn   = (wave >> 2) * 64;

  const v8f vzero = {0.f,0.f,0.f,0.f,0.f,0.f,0.f,0.f};
  v8f acc[2][4];
  #pragma unroll
  for (int i = 0; i < 2; ++i)
    #pragma unroll
    for (int j = 0; j < 4; ++j) acc[i][j] = vzero;

#if HAVE_TDM
  const unsigned as_off = (unsigned)(unsigned long long)(void*)&As[0][0];
  const unsigned bs_off = (unsigned)(unsigned long long)(void*)&Bs[0][0];
#else
  const int srow = tid >> 1;
  const int scol = (tid & 1) * 16;
#endif

  for (int k0 = 0; k0 < K; k0 += 32) {
    __syncthreads();
#if HAVE_TDM
    if (wave == 0) {
      tdm_load_2d(as_off, A, m0, k0, M, K, 128, 32);
      tdm_load_2d(bs_off, W, n0, k0, N, K, 128, 32);
      __builtin_amdgcn_s_wait_tensorcnt(0);
    }
#else
    {
      const long grow = m0 + srow;
      v8h a0 = {0,0,0,0,0,0,0,0}, a1 = {0,0,0,0,0,0,0,0};
      if (grow < M) {
        const f16* ap = A + grow * (long)K + k0 + scol;
        a0 = *(const v8h*)ap; a1 = *(const v8h*)(ap + 8);
      }
      for (int i = 0; i < 8; ++i) { As[srow][scol+i] = a0[i]; As[srow][scol+8+i] = a1[i]; }
      const long wrow = n0 + srow;
      v8h b0 = {0,0,0,0,0,0,0,0}, b1 = {0,0,0,0,0,0,0,0};
      if (wrow < N) {
        const f16* wp = W + wrow * (long)K + k0 + scol;
        b0 = *(const v8h*)wp; b1 = *(const v8h*)(wp + 8);
      }
      for (int i = 0; i < 8; ++i) { Bs[srow][scol+i] = b0[i]; Bs[srow][scol+8+i] = b1[i]; }
    }
#endif
    __syncthreads();

    const int fr = lane & 15;
    const int kb = (lane >> 4) * 8;
    v16h af0, af1, bf0, bf1, bf2, bf3;
    LOAD_FRAG(af0, As, wm + fr,      kb);
    LOAD_FRAG(af1, As, wm + 16 + fr, kb);
    LOAD_FRAG(bf0, Bs, wn + fr,      kb);
    LOAD_FRAG(bf1, Bs, wn + 16 + fr, kb);
    LOAD_FRAG(bf2, Bs, wn + 32 + fr, kb);
    LOAD_FRAG(bf3, Bs, wn + 48 + fr, kb);

    acc[0][0] = __builtin_amdgcn_wmma_f32_16x16x32_f16(false, af0, false, bf0, (short)0, acc[0][0], false, false);
    acc[0][1] = __builtin_amdgcn_wmma_f32_16x16x32_f16(false, af0, false, bf1, (short)0, acc[0][1], false, false);
    acc[0][2] = __builtin_amdgcn_wmma_f32_16x16x32_f16(false, af0, false, bf2, (short)0, acc[0][2], false, false);
    acc[0][3] = __builtin_amdgcn_wmma_f32_16x16x32_f16(false, af0, false, bf3, (short)0, acc[0][3], false, false);
    acc[1][0] = __builtin_amdgcn_wmma_f32_16x16x32_f16(false, af1, false, bf0, (short)0, acc[1][0], false, false);
    acc[1][1] = __builtin_amdgcn_wmma_f32_16x16x32_f16(false, af1, false, bf1, (short)0, acc[1][1], false, false);
    acc[1][2] = __builtin_amdgcn_wmma_f32_16x16x32_f16(false, af1, false, bf2, (short)0, acc[1][2], false, false);
    acc[1][3] = __builtin_amdgcn_wmma_f32_16x16x32_f16(false, af1, false, bf3, (short)0, acc[1][3], false, false);
  }

  const int nf = lane & 15;
  const int mh = (lane >> 4) << 3;
  #pragma unroll
  for (int i = 0; i < 2; ++i) {
    #pragma unroll
    for (int j = 0; j < 4; ++j) {
      const int n = n0 + wn + j * 16 + nf;
      const int rowb = m0 + wm + i * 16 + mh;
      #pragma unroll
      for (int e = 0; e < 8; ++e) {
        const int row = rowb + e;
        if (row < M) {
          float v = acc[i][j][e];
          if (bias) v += bias[n];
          if (res)  v += res[(long)row * N + n];
          if (relu) v = fmaxf(v, 0.f);
          if (Cf) Cf[(long)row * N + n] = v;
          if (Ch) Ch[(long)row * N + n] = (f16)v;
        }
      }
    }
  }
}

// ---------------------------------------------------------------------------
// Block reductions (blockDim == 256 -> 8 wave32 waves)
// ---------------------------------------------------------------------------
__device__ __forceinline__ float blockSum(float v, float* red) {
  #pragma unroll
  for (int m = 16; m; m >>= 1) v += __shfl_xor(v, m, 32);
  if ((threadIdx.x & 31) == 0) red[threadIdx.x >> 5] = v;
  __syncthreads();
  float tot = 0.f;
  #pragma unroll
  for (int i = 0; i < 8; ++i) tot += red[i];
  __syncthreads();
  return tot;
}
__device__ __forceinline__ float blockMax(float v, float* red) {
  #pragma unroll
  for (int m = 16; m; m >>= 1) v = fmaxf(v, __shfl_xor(v, m, 32));
  if ((threadIdx.x & 31) == 0) red[threadIdx.x >> 5] = v;
  __syncthreads();
  float tot = -3.4e38f;
  #pragma unroll
  for (int i = 0; i < 8; ++i) tot = fmaxf(tot, red[i]);
  __syncthreads();
  return tot;
}

// ---------------------------------------------------------------------------
// Embedding gather (emb f32 -> f16 rows), rows = (b,t)
// ---------------------------------------------------------------------------
__global__ __launch_bounds__(256) void k_gather(const int* __restrict__ tgt,
                                                const float* __restrict__ emb,
                                                f16* __restrict__ out) {
  const int r = blockIdx.x;
  const int b = r / TI, t = r % TI;
  const int tok = tgt[b * T_FULL + t];
  for (int e = threadIdx.x; e < D_MODEL; e += 256)
    out[(long)r * D_MODEL + e] = (f16)emb[(long)tok * D_MODEL + e];
}

// ---------------------------------------------------------------------------
// Sinusoidal positional encoding add (+ f16 mirror)
// ---------------------------------------------------------------------------
__global__ __launch_bounds__(256) void k_pe_add(float* __restrict__ x,
                                                f16* __restrict__ x16) {
  const int r = blockIdx.x;
  const int t = r % TI;
  const float c = -0.017988946f;   // -ln(10000)/512
  for (int d = threadIdx.x; d < D_MODEL; d += 256) {
    const int i2 = (d >> 1) * 2;
    const float ang = (float)t * __expf(c * (float)i2);
    const float pe = (d & 1) ? cosf(ang) : sinf(ang);
    const float v = x[(long)r * D_MODEL + d] + pe;
    x[(long)r * D_MODEL + d] = v;
    x16[(long)r * D_MODEL + d] = (f16)v;
  }
}

// ---------------------------------------------------------------------------
// LayerNorm over D=512, one row per block, f32 + f16 outputs
// ---------------------------------------------------------------------------
__global__ __launch_bounds__(256) void k_ln(const float* __restrict__ X,
                                            const float* __restrict__ g,
                                            const float* __restrict__ bt,
                                            float* __restrict__ Y,
                                            f16* __restrict__ Yh) {
  __shared__ float red[8];
  const int r = blockIdx.x, tid = threadIdx.x;
  const float* xr = X + (long)r * D_MODEL;
  const float v0 = xr[tid], v1 = xr[tid + 256];
  const float mean = blockSum(v0 + v1, red) * (1.f / 512.f);
  const float d0 = v0 - mean, d1 = v1 - mean;
  const float var = blockSum(d0 * d0 + d1 * d1, red) * (1.f / 512.f);
  const float rinv = rsqrtf(var + 1e-5f);
  const float y0 = d0 * rinv * g[tid]       + bt[tid];
  const float y1 = d1 * rinv * g[tid + 256] + bt[tid + 256];
  Y[(long)r * D_MODEL + tid]        = y0;
  Y[(long)r * D_MODEL + tid + 256]  = y1;
  Yh[(long)r * D_MODEL + tid]       = (f16)y0;
  Yh[(long)r * D_MODEL + tid + 256] = (f16)y1;
}

// ---------------------------------------------------------------------------
// MHA core: softmax(QK^T/8 + mask) V. One wave per (q, h, b). dh = 64.
// ---------------------------------------------------------------------------
__global__ __launch_bounds__(32) void k_attn(
    const float* __restrict__ Q, const float* __restrict__ Km,
    const float* __restrict__ Vm, float* __restrict__ O,
    const int* __restrict__ toks, int tokStride,
    int Lq, int Lk, int causal)
{
  __shared__ float p[S_LEN + 16];
  const int qi = blockIdx.x, h = blockIdx.y, b = blockIdx.z;
  const int lane = threadIdx.x;
  const float* qr = Q + ((long)(b * Lq + qi) * D_MODEL) + h * D_HEAD;

  float lmax = -3.4e38f;
  for (int k = lane; k < Lk; k += 32) {
    float s;
    const bool masked = (causal && k > qi) || (toks && toks[b * tokStride + k] == 0);
    if (masked) s = -1e9f;
    else {
      const float* kr = Km + ((long)(b * Lk + k) * D_MODEL) + h * D_HEAD;
      float acc = 0.f;
      #pragma unroll 8
      for (int d = 0; d < D_HEAD; ++d) acc += qr[d] * kr[d];
      s = acc * 0.125f;                 // 1/sqrt(64)
    }
    p[k] = s;
    lmax = fmaxf(lmax, s);
  }
  #pragma unroll
  for (int m = 16; m; m >>= 1) lmax = fmaxf(lmax, __shfl_xor(lmax, m, 32));
  __syncthreads();

  float lsum = 0.f;
  for (int k = lane; k < Lk; k += 32) {
    const float e = __expf(p[k] - lmax);
    p[k] = e;
    lsum += e;
  }
  #pragma unroll
  for (int m = 16; m; m >>= 1) lsum += __shfl_xor(lsum, m, 32);
  const float inv = 1.f / lsum;
  __syncthreads();

  float o0 = 0.f, o1 = 0.f;            // lane owns d = lane, lane+32
  for (int k = 0; k < Lk; ++k) {
    const float* vr = Vm + ((long)(b * Lk + k) * D_MODEL) + h * D_HEAD;
    const float pk = p[k] * inv;
    o0 += pk * vr[lane];
    o1 += pk * vr[lane + 32];
  }
  float* orow = O + ((long)(b * Lq + qi) * D_MODEL) + h * D_HEAD;
  orow[lane] = o0;
  orow[lane + 32] = o1;
}

// ---------------------------------------------------------------------------
// Copy dec f16 into first half of concat buffer [dec | context]
// ---------------------------------------------------------------------------
__global__ __launch_bounds__(256) void k_cat_dec(const f16* __restrict__ dec16,
                                                 f16* __restrict__ xcat) {
  const int r = blockIdx.x;
  for (int d = threadIdx.x; d < D_MODEL; d += 256)
    xcat[(long)r * (2 * D_MODEL) + d] = dec16[(long)r * D_MODEL + d];
}

// ---------------------------------------------------------------------------
// One coverage/pointer timestep (sequential through cov). Heavy matmuls
// (enc_feat, dec@ws.T+battn, vocab projection) are hoisted to WMMA GEMMs.
// ---------------------------------------------------------------------------
__global__ __launch_bounds__(256) void k_cov_step(
    int t,
    const float* __restrict__ enc_feat, const float* __restrict__ dec_ws,
    const float* __restrict__ wc, const float* __restrict__ vvec,
    const float* __restrict__ enc, const float* __restrict__ dec,
    const float* __restrict__ pgen_w, const float* __restrict__ pgen_b,
    float* __restrict__ cov, float* __restrict__ attn_all,
    float* __restrict__ ctx, f16* __restrict__ xcat,
    float* __restrict__ pgen, float* __restrict__ loss)
{
  __shared__ float sc[S_LEN];
  __shared__ float red[8];
  const int b = blockIdx.x, tid = threadIdx.x;
  const long rrow = (long)b * TI + t;
  const float* dw = dec_ws + rrow * D_MODEL;

  for (int s = tid; s < S_LEN; s += 256) {
    const float* ef = enc_feat + ((long)b * S_LEN + s) * D_MODEL;
    const float cv = cov[b * S_LEN + s];
    float acc = 0.f;
    for (int d = 0; d < D_MODEL; ++d)
      acc += tanhf(ef[d] + dw[d] + cv * wc[d]) * vvec[d];
    sc[s] = acc;
  }
  __syncthreads();

  float lm = -3.4e38f;
  for (int s = tid; s < S_LEN; s += 256) lm = fmaxf(lm, sc[s]);
  lm = blockMax(lm, red);
  float ls = 0.f;
  for (int s = tid; s < S_LEN; s += 256) {
    const float e = __expf(sc[s] - lm);
    sc[s] = e;
    ls += e;
  }
  ls = blockSum(ls, red);
  const float inv = 1.f / ls;

  float lloss = 0.f;
  for (int s = tid; s < S_LEN; s += 256) {
    const float a = sc[s] * inv;
    sc[s] = a;
    const float cv = cov[b * S_LEN + s];
    lloss += fminf(a, cv);
    attn_all[rrow * S_LEN + s] = a;
    cov[b * S_LEN + s] = cv + a;
  }
  lloss = blockSum(lloss, red);
  if (tid == 0) atomicAdd(loss, lloss * (1.0f / (float)(B_SZ * TI)));
  __syncthreads();

  const float* drow = dec + rrow * D_MODEL;
  float pdot = 0.f;
  for (int d = tid; d < D_MODEL; d += 256) {
    float c = 0.f;
    for (int s = 0; s < S_LEN; ++s)
      c += sc[s] * enc[((long)b * S_LEN + s) * D_MODEL + d];
    ctx[rrow * D_MODEL + d] = c;
    xcat[rrow * (2 * D_MODEL) + D_MODEL + d] = (f16)c;
    const float dv = drow[d];
    pdot += c * pgen_w[d] + dv * (pgen_w[D_MODEL + d] + pgen_w[2 * D_MODEL + d]);
  }
  pdot = blockSum(pdot, red);
  if (tid == 0) pgen[rrow] = 1.f / (1.f + __expf(-(pdot + pgen_b[0])));
}

// ---------------------------------------------------------------------------
// Final: per-row softmax over V, scale by p_gen, scatter-add (1-p_gen)*attn
// ---------------------------------------------------------------------------
__global__ __launch_bounds__(256) void k_final(
    float* __restrict__ out, const float* __restrict__ pgen,
    const float* __restrict__ attn_all, const int* __restrict__ srctok)
{
  __shared__ float red[8];
  const long r = blockIdx.x;
  float* row = out + r * V_SZ;
  const int tid = threadIdx.x;

  float lm = -3.4e38f;
  for (int i = tid; i < V_SZ; i += 256) lm = fmaxf(lm, row[i]);
  lm = blockMax(lm, red);
  float ls = 0.f;
  for (int i = tid; i < V_SZ; i += 256) ls += __expf(row[i] - lm);
  ls = blockSum(ls, red);

  const float pg = pgen[r];
  const float scale = pg / ls;
  for (int i = tid; i < V_SZ; i += 256) row[i] = __expf(row[i] - lm) * scale;
  __syncthreads();

  const int b = (int)(r / TI);
  const float q = 1.f - pg;
  for (int s = tid; s < S_LEN; s += 256) {
    const int tok = srctok[b * S_LEN + s];
    atomicAdd(&row[tok], q * attn_all[r * S_LEN + s]);
  }
}

// ---------------------------------------------------------------------------
// Host orchestration
// ---------------------------------------------------------------------------
extern "C" void kernel_launch(void* const* d_in, const int* in_sizes, int n_in,
                              void* d_out, int out_size, void* d_ws, size_t ws_size,
                              hipStream_t stream) {
  (void)in_sizes; (void)n_in; (void)out_size; (void)ws_size;

  const int*   tgt  = (const int*)  d_in[0];
  const float* enc  = (const float*)d_in[1];
  const int*   srct = (const int*)  d_in[2];
  auto P = [&](int i) { return (const float*)d_in[i]; };
  // emb=3, proj_w=4, proj_b=5 ; layer base = 6+26l ; cov 84-88 ; ptr 89-92

  char* wsp = (char*)d_ws;
  size_t off = 0;
  auto alloc = [&](size_t bytes) -> void* {
    void* p = wsp + off;
    off = (off + bytes + 255) & ~(size_t)255;
    return p;
  };
  auto toh = [&](const float* src, long n) -> f16* {
    f16* dst = (f16*)alloc((size_t)n * sizeof(f16));
    k_convert<<<2048, 256, 0, stream>>>(src, dst, n);
    return dst;
  };
  auto gemm = [&](const f16* A, const f16* W, const float* bias, const float* res,
                  float* Cf, f16* Ch, int M, int N, int K, int relu) {
    dim3 g(N / 128, (M + 127) / 128);
    k_gemm<<<g, 256, 0, stream>>>(A, W, bias, res, Cf, Ch, M, N, K, relu);
  };
  auto gemm_tdm = [&](const f16* A, const f16* W, const float* bias, const float* res,
                      float* Cf, f16* Ch, int M, int N, int K, int relu) {
    dim3 g(N / 128, (M + 127) / 128);
    k_gemm_tdm<<<g, 256, 0, stream>>>(A, W, bias, res, Cf, Ch, M, N, K, relu);
  };

  const int M = M_ROWS;
  const int MS = B_SZ * S_LEN;                       // 6400 encoder rows

  float* x        = (float*)alloc((size_t)M * D_MODEL * 4);
  f16*   x16      = (f16*)  alloc((size_t)M * D_MODEL * 2);
  float* y        = (float*)alloc((size_t)M * D_MODEL * 4);
  float* qf       = (float*)alloc((size_t)M * D_MODEL * 4);
  float* kfs      = (float*)alloc((size_t)M * D_MODEL * 4);
  float* vfs      = (float*)alloc((size_t)M * D_MODEL * 4);
  float* kc       = (float*)alloc((size_t)MS * D_MODEL * 4);
  float* vc       = (float*)alloc((size_t)MS * D_MODEL * 4);
  float* att_o    = (float*)alloc((size_t)M * D_MODEL * 4);
  f16*   att_o16  = (f16*)  alloc((size_t)M * D_MODEL * 2);
  f16*   h16      = (f16*)  alloc((size_t)M * 2048 * 2);
  f16*   enc16    = (f16*)  alloc((size_t)MS * D_MODEL * 2);
  f16*   act16    = (f16*)  alloc((size_t)M * D_MODEL * 2);
  float* enc_feat = (float*)alloc((size_t)MS * D_MODEL * 4);
  float* dec_wsb  = (float*)alloc((size_t)M * D_MODEL * 4);
  float* ctx      = (float*)alloc((size_t)M * D_MODEL * 4);
  f16*   xcat     = (f16*)  alloc((size_t)M * 2 * D_MODEL * 2);
  float* attn_all = (float*)alloc((size_t)M * S_LEN * 4);
  float* covb     = (float*)alloc((size_t)B_SZ * S_LEN * 4);
  float* pgenb    = (float*)alloc((size_t)M * 4);

  f16* proj_w16 = toh(P(4), 512L * 512);
  const f16 *saW[3][4], *caW[3][4], *w1h[3], *w2h[3];
  for (int l = 0; l < 3; ++l) {
    const int base = 6 + 26 * l;
    for (int j = 0; j < 4; ++j) saW[l][j] = toh(P(base + 2 * j), 512L * 512);
    for (int j = 0; j < 4; ++j) caW[l][j] = toh(P(base + 8 + 2 * j), 512L * 512);
    w1h[l] = toh(P(base + 16), 2048L * 512);
    w2h[l] = toh(P(base + 18), 512L * 2048);
  }
  f16* wh16   = toh(P(84), 512L * 512);
  f16* wsc16  = toh(P(85), 512L * 512);
  f16* outw16 = toh(P(91), (long)V_SZ * 1024);
  k_convert<<<2048, 256, 0, stream>>>(enc, enc16, (long)MS * D_MODEL);

  // ---- embedding + projection + positional encoding ----
  k_gather<<<M, 256, 0, stream>>>(tgt, P(3), act16);
  gemm(act16, proj_w16, P(5), nullptr, x, nullptr, M, 512, 512, 0);
  k_pe_add<<<M, 256, 0, stream>>>(x, x16);

  // ---- decoder layers ----
  for (int l = 0; l < 3; ++l) {
    const int base = 6 + 26 * l;
    gemm(x16, saW[l][0], P(base + 1), nullptr, qf,  nullptr, M, 512, 512, 0);
    gemm(x16, saW[l][1], P(base + 3), nullptr, kfs, nullptr, M, 512, 512, 0);
    gemm(x16, saW[l][2], P(base + 5), nullptr, vfs, nullptr, M, 512, 512, 0);
    k_attn<<<dim3(TI, N_HEAD, B_SZ), 32, 0, stream>>>(qf, kfs, vfs, att_o,
                                                      tgt, T_FULL, TI, TI, 1);
    k_convert<<<2048, 256, 0, stream>>>(att_o, att_o16, (long)M * D_MODEL);
    gemm(att_o16, saW[l][3], P(base + 7), x, y, nullptr, M, 512, 512, 0);
    k_ln<<<M, 256, 0, stream>>>(y, P(base + 20), P(base + 21), x, x16);

    gemm(x16,   caW[l][0], P(base + 9),  nullptr, qf, nullptr, M,  512, 512, 0);
    gemm(enc16, caW[l][1], P(base + 11), nullptr, kc, nullptr, MS, 512, 512, 0);
    gemm(enc16, caW[l][2], P(base + 13), nullptr, vc, nullptr, MS, 512, 512, 0);
    k_attn<<<dim3(TI, N_HEAD, B_SZ), 32, 0, stream>>>(qf, kc, vc, att_o,
                                                      srct, S_LEN, TI, S_LEN, 0);
    k_convert<<<2048, 256, 0, stream>>>(att_o, att_o16, (long)M * D_MODEL);
    gemm(att_o16, caW[l][3], P(base + 15), x, y, nullptr, M, 512, 512, 0);
    k_ln<<<M, 256, 0, stream>>>(y, P(base + 22), P(base + 23), x, x16);

    gemm(x16, w1h[l], P(base + 17), nullptr, nullptr, h16, M, 2048, 512, 1);
    gemm(h16, w2h[l], P(base + 19), x, y, nullptr, M, 512, 2048, 0);
    k_ln<<<M, 256, 0, stream>>>(y, P(base + 24), P(base + 25), x, x16);
  }

  // ---- coverage/pointer head ----
  gemm_tdm(enc16, wh16, nullptr, nullptr, enc_feat, nullptr, MS, 512, 512, 0);
  gemm(x16, wsc16, P(87), nullptr, dec_wsb, nullptr, M, 512, 512, 0);
  k_zero<<<32, 256, 0, stream>>>(covb, (long)B_SZ * S_LEN);
  float* lossp = (float*)d_out + (long)M * V_SZ;
  k_zero<<<1, 32, 0, stream>>>(lossp, 1);
  k_cat_dec<<<M, 256, 0, stream>>>(x16, xcat);

  for (int t = 0; t < TI; ++t) {
    k_cov_step<<<B_SZ, 256, 0, stream>>>(t, enc_feat, dec_wsb, P(86), P(88),
                                         enc, x, P(89), P(90),
                                         covb, attn_all, ctx, xcat, pgenb, lossp);
  }

  // ---- batched vocab projection (dominant GEMM) straight into d_out ----
  gemm(xcat, outw16, P(92), nullptr, (float*)d_out, nullptr, M, V_SZ, 1024, 0);
  k_final<<<M, 256, 0, stream>>>((float*)d_out, pgenb, attn_all, srct);
}
